// DetMetaDynamics_62045097558100
// MI455X (gfx1250) — matrix operand order repaired and checked by reading, not compiled, not run on hardware
//
#include <hip/hip_runtime.h>
#include <math.h>

typedef __attribute__((ext_vector_type(16))) _Float16 v16h;
typedef __attribute__((ext_vector_type(8)))  _Float16 v8h;
typedef __attribute__((ext_vector_type(8)))  float    v8f;

// ---------------------------------------------------------------------------
// CDNA5 wave32 WMMA 16x16x32 f16->f32.
// A 16x32 f16: lane L (row m=L&15, half h=L>>4) holds K = [h*8, h*8+8) and
//   [16+h*8, 16+h*8+8)  -> two contiguous 16B chunks when K is memory-contiguous.
// B 32x16 f16: lane L (col n=L&15, half h) holds K = [h*8, h*8+8) and
//   [16+h*8, 16+h*8+8)  -> two contiguous 16B chunks (same as A).
// C/D 16x16 f32: VGPR r: row = r + 8*(L>>4), col = L&15.
// ---------------------------------------------------------------------------
__device__ __forceinline__ v16h cat8(v8h lo, v8h hi) {
    return __builtin_shufflevector(lo, hi, 0, 1, 2, 3, 4, 5, 6, 7,
                                   8, 9, 10, 11, 12, 13, 14, 15);
}
__device__ __forceinline__ v8f wmma_f16(v16h a, v16h b, v8f c) {
    return __builtin_amdgcn_wmma_f32_16x16x32_f16(false, a, false, b, (short)0, c,
                                                  false, false);
}
// A fragment: 32 K-contiguous halves at xa (16B aligned)
__device__ __forceinline__ v16h frag_a(const _Float16* xa, int h) {
    v8h lo = *(const v8h*)(xa + h * 8);
    v8h hi = *(const v8h*)(xa + 16 + h * 8);
    return cat8(lo, hi);
}
// B fragment: 32 K-contiguous halves at wa (16B aligned)
__device__ __forceinline__ v16h frag_b(const _Float16* wa, int h) {
    v8h lo = *(const v8h*)(wa + h * 8);
    v8h hi = *(const v8h*)(wa + 16 + h * 8);
    return cat8(lo, hi);
}

// ---------------------------------------------------------------------------
// Pack NCHW f32 activations -> zero-halo padded NHWC f16, channels padded.
// dst: [NI][H+2p][W+2p][Cp]; halos pre-zeroed by memset.
// ---------------------------------------------------------------------------
__global__ void pack_act_kernel(const float* __restrict__ src, long imgStride,
                                _Float16* __restrict__ dst,
                                int NI, int Cin, int Cp, int H, int pad)
{
    const int W = H, HP = H + 2 * pad, WP = W + 2 * pad;
    long total = (long)NI * H * W * Cp;
    long idx = (long)blockIdx.x * 256 + threadIdx.x;
    if (idx >= total) return;
    int c = (int)(idx % Cp);
    long t = idx / Cp;
    int x = (int)(t % W); t /= W;
    int y = (int)(t % H); t /= H;
    int ni = (int)t;
    float v = (c < Cin) ? src[(size_t)ni * imgStride + (size_t)c * H * W + y * W + x]
                        : 0.f;
    dst[(((size_t)ni * HP + y + pad) * WP + x + pad) * Cp + c] = (_Float16)v;
}

// [Cout][Cin][T] f32 -> [Cout][T][Cp] f16 (K-contiguous per (co,tap))
__global__ void pack_w_kernel(const float* __restrict__ w, _Float16* __restrict__ wp,
                              int Cout, int Cin, int Cp, int T)
{
    int idx = blockIdx.x * 256 + threadIdx.x;
    int total = Cout * T * Cp;
    if (idx >= total) return;
    int c = idx % Cp;
    int t = (idx / Cp) % T;
    int co = idx / (Cp * T);
    wp[idx] = (c < Cin) ? (_Float16)w[((size_t)co * Cin + c) * T + t] : (_Float16)0.f;
}

// eWin [16][8192] f32 -> transposed f16 [8192][32] (K padded 16->32)
__global__ void pack_ewin_kernel(const float* __restrict__ w, _Float16* __restrict__ wp)
{
    int idx = blockIdx.x * 256 + threadIdx.x;   // 8192*32
    int k = idx & 31, n = idx >> 5;
    wp[idx] = (k < 16) ? (_Float16)w[(size_t)k * 8192 + n] : (_Float16)0.f;
}

// zall [t][b][16] f32 -> z_flat f16 [3200][32] (row m = b*100+t, K padded)
__global__ void pack_z_kernel(const float* __restrict__ zall, _Float16* __restrict__ zf)
{
    int idx = blockIdx.x * 256 + threadIdx.x;   // 3200*32
    int k = idx & 31, m = idx >> 5;
    int b = m / 100, t = m % 100;
    zf[idx] = (k < 16) ? (_Float16)zall[((size_t)t * 32 + b) * 16 + k] : (_Float16)0.f;
}

// ---------------------------------------------------------------------------
// 5x5 stride-2 SAME conv, implicit GEMM on WMMA. All loads unconditional.
// Each wave computes a 16(M) x 32(N) C-tile: one A fragment feeds two WMMAs.
// xp: [NI][H+4][W+4][Cp] f16 (zero halo); wp: [Cout][25][Cp] f16
// yp: [NI][OH+2po][OW+2po][Cout] f16 (interior written; halo pre-zeroed)
// act: 0 = leaky-relu(0.1), 1 = tanh
// ---------------------------------------------------------------------------
__global__ __launch_bounds__(256) void conv5s2_kernel(
    const _Float16* __restrict__ xp, const _Float16* __restrict__ wp,
    const float* __restrict__ bias, _Float16* __restrict__ yp,
    int NI, int Cp, int H, int Cout, int po, int act)
{
    const int W = H, OH = H >> 1, OW = W >> 1, OP = OH * OW;
    const int WP = W + 4;
    const int OHP = OH + 2 * po, OWP = OW + 2 * po;
    const int M = NI * OP;
    int wave = threadIdx.x >> 5, lane = threadIdx.x & 31;
    int mt = blockIdx.x * 8 + wave;
    int m0 = mt * 16;
    if (m0 >= M) return;                         // wave-uniform
    int rm = m0 + (lane & 15);
    int rmc = rm < M ? rm : M - 1;               // clamp: loads stay in-bounds
    int img = rmc / OP, rr = rmc % OP, oy = rr / OW, ox = rr % OW;
    int cn0 = blockIdx.y * 32 + (lane & 15);     // Cout multiple of 32
    int cn1 = cn0 + 16;
    int h = lane >> 4;

    const _Float16* xrow = xp + (((size_t)img * (H + 4) + 2 * oy) * WP + 2 * ox) * Cp;
    const _Float16* wcol0 = wp + (size_t)cn0 * 25 * Cp;
    const _Float16* wcol1 = wp + (size_t)cn1 * 25 * Cp;

    v8f acc0 = {}, acc1 = {};
    for (int tap = 0; tap < 25; ++tap) {
        int ky = tap / 5, kx = tap % 5;
        const _Float16* xa0 = xrow + ((size_t)ky * WP + kx) * Cp;
        const _Float16* wa0 = wcol0 + (size_t)tap * Cp;
        const _Float16* wa1 = wcol1 + (size_t)tap * Cp;
        __builtin_prefetch(xa0 + (size_t)WP * Cp, 0, 0);
        for (int kc = 0; kc < Cp; kc += 32) {
            v16h a  = frag_a(xa0 + kc, h);
            v16h b0 = frag_b(wa0 + kc, h);
            v16h b1 = frag_b(wa1 + kc, h);
            acc0 = wmma_f16(a, b0, acc0);
            acc1 = wmma_f16(a, b1, acc1);
        }
    }
    float bv0 = bias[cn0], bv1 = bias[cn1];
#pragma unroll
    for (int r = 0; r < 8; ++r) {
        int cm = m0 + r + 8 * h;
        if (cm < M) {
            int im2 = cm / OP, q = cm % OP, oy2 = q / OW, ox2 = q % OW;
            size_t base = (((size_t)im2 * OHP + oy2 + po) * OWP + ox2 + po) * Cout;
            float v0 = acc0[r] + bv0;
            float v1 = acc1[r] + bv1;
            v0 = (act == 0) ? (v0 > 0.f ? v0 : 0.1f * v0) : tanhf(v0);
            v1 = (act == 0) ? (v1 > 0.f ? v1 : 0.1f * v1) : tanhf(v1);
            yp[base + cn0] = (_Float16)v0;
            yp[base + cn1] = (_Float16)v1;
        }
    }
}

// ---------------------------------------------------------------------------
// Global-avg-pool (4x4) + linear heads; conv3 output is NHWC f16 [NI][16][128].
// ---------------------------------------------------------------------------
__global__ void domain_head_kernel(const _Float16* __restrict__ c3,
                                   const float* __restrict__ Wo,
                                   const float* __restrict__ bo,
                                   float* __restrict__ zk)
{
    __shared__ float pooled[128];
    int ni = blockIdx.x, t = threadIdx.x;   // 128 threads
    const _Float16* p = c3 + (size_t)ni * 2048 + t;
    float s = 0.f;
#pragma unroll
    for (int i = 0; i < 16; ++i) s += (float)p[i * 128];
    pooled[t] = s * (1.f / 16.f);
    __syncthreads();
    if (t < 16) {
        float z = bo[t];
        for (int c = 0; c < 128; ++c) z += pooled[c] * Wo[c * 16 + t];
        zk[ni * 16 + t] = z;
    }
}

__global__ void zc_kernel(const float* __restrict__ zk, float* __restrict__ zc)
{
    int i = threadIdx.x;    // one block, 512 threads
    if (i < 512) {
        int b = i >> 4, l = i & 15;
        float s = 0.f;
#pragma unroll
        for (int k = 0; k < 8; ++k) s += zk[(b * 8 + k) * 16 + l];
        zc[i] = s * 0.125f;
    }
}

__global__ void state_head_kernel(const _Float16* __restrict__ c3,
                                  const float* __restrict__ Wmu,
                                  const float* __restrict__ bmu,
                                  const float* __restrict__ Wvar,
                                  const float* __restrict__ bvar,
                                  float* __restrict__ out_mu,
                                  float* __restrict__ out_var,
                                  float* __restrict__ zall)
{
    __shared__ float pooled[128];
    int b = blockIdx.x, t = threadIdx.x;
    const _Float16* p = c3 + (size_t)b * 2048 + t;
    float s = 0.f;
#pragma unroll
    for (int i = 0; i < 16; ++i) s += (float)p[i * 128];
    pooled[t] = s * (1.f / 16.f);
    __syncthreads();
    if (t < 16) {
        float mu = bmu[t];
        for (int c = 0; c < 128; ++c) mu += pooled[c] * Wmu[c * 16 + t];
        out_mu[b * 16 + t] = mu;
        zall[b * 16 + t] = mu;          // z_0 = mu_0
    } else if (t < 32) {
        int l = t - 16;
        float v = bvar[l];
        for (int c = 0; c < 128; ++c) v += pooled[c] * Wvar[c * 16 + l];
        out_var[b * 16 + l] = (v > 20.f) ? v : log1pf(expf(v));
    }
}

// ---------------------------------------------------------------------------
// Sequential gated transition: 1 persistent block, weights in LDS (latency-bound).
// zall layout [t][b][l].
// ---------------------------------------------------------------------------
__global__ __launch_bounds__(256) void transition_kernel(
    const float* __restrict__ Wg1, const float* __restrict__ bg1,
    const float* __restrict__ Wg2, const float* __restrict__ bg2,
    const float* __restrict__ Wp1, const float* __restrict__ bp1,
    const float* __restrict__ Wp2, const float* __restrict__ bp2,
    const float* __restrict__ Wz,  const float* __restrict__ bz,
    const float* __restrict__ zc,  float* __restrict__ zall)
{
    __shared__ float sWg1[2048], sWp1[2048], sWg2[1024], sWp2[1024], sWz[256];
    __shared__ float sbg1[64], sbp1[64], sbg2[16], sbp2[16], sbz[16];
    __shared__ float zp[512], zcs[512], hg[2048], hp[2048], zn[512];
    int t = threadIdx.x;
    for (int i = t; i < 2048; i += 256) { sWg1[i] = Wg1[i]; sWp1[i] = Wp1[i]; }
    for (int i = t; i < 1024; i += 256) { sWg2[i] = Wg2[i]; sWp2[i] = Wp2[i]; }
    for (int i = t; i < 256;  i += 256) { sWz[i] = Wz[i]; }
    if (t < 64) { sbg1[t] = bg1[t]; sbp1[t] = bp1[t]; }
    if (t < 16) { sbg2[t] = bg2[t]; sbp2[t] = bp2[t]; sbz[t] = bz[t]; }
    for (int i = t; i < 512; i += 256) { zp[i] = zall[i]; zcs[i] = zc[i]; }
    __syncthreads();

    for (int step = 1; step < 100; ++step) {
        for (int e = t; e < 2048; e += 256) {
            int b = e >> 6, j = e & 63;
            float sg = sbg1[j], sp = sbp1[j];
#pragma unroll
            for (int i = 0; i < 16; ++i) {
                float zv = zp[b * 16 + i];
                sg += zv * sWg1[i * 64 + j];
                sp += zv * sWp1[i * 64 + j];
            }
#pragma unroll
            for (int i = 0; i < 16; ++i) {
                float zv = zcs[b * 16 + i];
                sg += zv * sWg1[(16 + i) * 64 + j];
                sp += zv * sWp1[(16 + i) * 64 + j];
            }
            hg[e] = sg > 0.f ? sg : 0.f;
            hp[e] = sp > 0.f ? sp : 0.f;
        }
        __syncthreads();
        for (int e = t; e < 512; e += 256) {
            int b = e >> 4, l = e & 15;
            float g = sbg2[l], pr = sbp2[l], lin = sbz[l];
            for (int j = 0; j < 64; ++j) {
                g  += hg[b * 64 + j] * sWg2[j * 16 + l];
                pr += hp[b * 64 + j] * sWp2[j * 16 + l];
            }
#pragma unroll
            for (int i = 0; i < 16; ++i) lin += zp[b * 16 + i] * sWz[i * 16 + l];
            g = 1.f / (1.f + expf(-g));
            float z = (1.f - g) * lin + g * pr;
            zn[e] = z;
            zall[(size_t)step * 512 + e] = z;
        }
        __syncthreads();
        for (int e = t; e < 512; e += 256) zp[e] = zn[e];
        __syncthreads();
    }
}

// ---------------------------------------------------------------------------
// Emission linear: zf16[3200][32] @ ewT[8192][32] + ebin -> padded NHWC
// h0p [3200][10][10][128] f16.  One A fragment, two N-tiles per wave.
// ---------------------------------------------------------------------------
__global__ __launch_bounds__(256) void lin_kernel(
    const _Float16* __restrict__ zf, const _Float16* __restrict__ wT,
    const float* __restrict__ bi, _Float16* __restrict__ h0p)
{
    const int M = 3200;
    int wave = threadIdx.x >> 5, lane = threadIdx.x & 31;
    int mt = blockIdx.x * 8 + wave;
    int m0 = mt * 16;
    if (m0 >= M) return;
    int rm = m0 + (lane & 15);
    int rmc = rm < M ? rm : M - 1;
    int cn0 = blockIdx.y * 32 + (lane & 15);
    int cn1 = cn0 + 16;
    int h = lane >> 4;

    v16h a  = frag_a(zf + (size_t)rmc * 32, h);
    v16h b0 = frag_b(wT + (size_t)cn0 * 32, h);
    v16h b1 = frag_b(wT + (size_t)cn1 * 32, h);
    v8f acc0 = {}, acc1 = {};
    acc0 = wmma_f16(a, b0, acc0);
    acc1 = wmma_f16(a, b1, acc1);

    float bv0 = bi[cn0], bv1 = bi[cn1];
    int co0 = cn0 >> 6, pix0 = cn0 & 63, py0 = (pix0 >> 3) + 1, px0 = (pix0 & 7) + 1;
    int co1 = cn1 >> 6, pix1 = cn1 & 63, py1 = (pix1 >> 3) + 1, px1 = (pix1 & 7) + 1;
#pragma unroll
    for (int r = 0; r < 8; ++r) {
        int cm = m0 + r + 8 * h;
        if (cm < M) {
            h0p[(((size_t)cm * 10 + py0) * 10 + px0) * 128 + co0] =
                (_Float16)(acc0[r] + bv0);
            h0p[(((size_t)cm * 10 + py1) * 10 + px1) * 128 + co1] =
                (_Float16)(acc1[r] + bv1);
        }
    }
}

// ---------------------------------------------------------------------------
// Transposed conv stride-2 k=4, decomposed into 4 parity classes (blockIdx.z),
// each a dense 2x2-tap conv over the 1-ring-padded NHWC input. No dead taps,
// no predication. Each wave computes a 16(M) x 32(N) C-tile.
// xp: [NB][H+2][W+2][Cp]; wp: [Cout][16][Cp]; yp: [NB][2H+2][2W+2][Cout]
// ---------------------------------------------------------------------------
__global__ __launch_bounds__(256) void convt_kernel(
    const _Float16* __restrict__ xp, const _Float16* __restrict__ wp,
    const float* __restrict__ bias, _Float16* __restrict__ yp,
    int NB, int Cp, int H, int Cout)
{
    const int W = H, WP = W + 2, OHP = 2 * H + 2, OWP = 2 * W + 2;
    const int MP = H * W;
    const int M = NB * MP;
    int pa = blockIdx.z >> 1, pb = blockIdx.z & 1;
    int wave = threadIdx.x >> 5, lane = threadIdx.x & 31;
    int mt = blockIdx.x * 8 + wave;
    int m0 = mt * 16;
    if (m0 >= M) return;
    int rm = m0 + (lane & 15);
    int rmc = rm < M ? rm : M - 1;
    int img = rmc / MP, q = rmc % MP, i = q / W, j = q % W;
    int cn0 = blockIdx.y * 32 + (lane & 15);
    int cn1 = cn0 + 16;
    int h = lane >> 4;

    const _Float16* xbase = xp + (size_t)img * (H + 2) * WP * Cp;
    const _Float16* wcol0 = wp + (size_t)cn0 * 16 * Cp;
    const _Float16* wcol1 = wp + (size_t)cn1 * 16 * Cp;

    v8f acc0 = {}, acc1 = {};
#pragma unroll
    for (int ty = 0; ty < 2; ++ty) {
        int iy = i + ty + pa;
        int ky = 2 * ty + pa;
#pragma unroll
        for (int tx = 0; tx < 2; ++tx) {
            int ix = j + tx + pb;
            int kx = 2 * tx + pb;
            const _Float16* xa0 = xbase + ((size_t)iy * WP + ix) * Cp;
            const _Float16* wa0 = wcol0 + (size_t)(ky * 4 + kx) * Cp;
            const _Float16* wa1 = wcol1 + (size_t)(ky * 4 + kx) * Cp;
            for (int kc = 0; kc < Cp; kc += 32) {
                v16h a  = frag_a(xa0 + kc, h);
                v16h b0 = frag_b(wa0 + kc, h);
                v16h b1 = frag_b(wa1 + kc, h);
                acc0 = wmma_f16(a, b0, acc0);
                acc1 = wmma_f16(a, b1, acc1);
            }
        }
    }
    float bv0 = bias[cn0], bv1 = bias[cn1];
#pragma unroll
    for (int r = 0; r < 8; ++r) {
        int cm = m0 + r + 8 * h;
        if (cm < M) {
            int im2 = cm / MP, q2 = cm % MP, i2 = q2 / W, j2 = q2 % W;
            int oy = 2 * i2 + pa + 1, ox = 2 * j2 + pb + 1;
            size_t base = (((size_t)im2 * OHP + oy) * OWP + ox) * Cout;
            float v0 = acc0[r] + bv0;
            float v1 = acc1[r] + bv1;
            v0 = v0 > 0.f ? v0 : 0.1f * v0;
            v1 = v1 > 0.f ? v1 : 0.1f * v1;
            yp[base + cn0] = (_Float16)v0;
            yp[base + cn1] = (_Float16)v1;
        }
    }
}

// ---------------------------------------------------------------------------
// Final 3x3 conv Cin=32 -> 1 + sigmoid (N=1 => VALU). Reads 1-ring-padded NHWC.
// x: [3200][34][34][32] f16; out f32 [3200][1024] == d_out head.
// ---------------------------------------------------------------------------
__global__ __launch_bounds__(256) void outconv_kernel(
    const _Float16* __restrict__ x, const float* __restrict__ w,
    const float* __restrict__ bias, float* __restrict__ out)
{
    __shared__ float sw[288];
    __shared__ float sb;
    for (int i = threadIdx.x; i < 288; i += 256) sw[i] = w[i];
    if (threadIdx.x == 0) sb = bias[0];
    __syncthreads();

    int idx = blockIdx.x * 256 + threadIdx.x;   // < 3,276,800
    int m = idx >> 10, pix = idx & 1023;
    int oy = pix >> 5, ox = pix & 31;
    const _Float16* xb = x + (size_t)m * 34 * 34 * 32;
    float s = sb;
#pragma unroll
    for (int ky = 0; ky < 3; ++ky) {
#pragma unroll
        for (int kx = 0; kx < 3; ++kx) {
            const v8h* p = (const v8h*)(xb + (((size_t)(oy + ky)) * 34 + ox + kx) * 32);
#pragma unroll
            for (int g = 0; g < 4; ++g) {
                v8h v = p[g];
#pragma unroll
                for (int e = 0; e < 8; ++e)
                    s += (float)v[e] * sw[(g * 8 + e) * 9 + ky * 3 + kx];
            }
        }
    }
    out[idx] = 1.f / (1.f + expf(-s));
}

// ---------------------------------------------------------------------------
// Launcher
// ---------------------------------------------------------------------------
enum {
    IN_X = 0, IN_D,
    P_DW1, P_DB1, P_DW2, P_DB2, P_DW3, P_DB3, P_DWO, P_DBO,
    P_IW1, P_IB1, P_IW2, P_IB2, P_IW3, P_IB3, P_IWMU, P_IBMU, P_IWVAR, P_IBVAR,
    P_WG1, P_BG1, P_WG2, P_BG2, P_WP1, P_BP1, P_WP2, P_BP2, P_WZ, P_BZ,
    P_EWIN, P_EBIN, P_ET1, P_EB1, P_ET2, P_EB2, P_EO, P_EBO
};

extern "C" void kernel_launch(void* const* d_in, const int* in_sizes, int n_in,
                              void* d_out, int out_size, void* d_ws, size_t ws_size,
                              hipStream_t stream) {
    (void)in_sizes; (void)n_in; (void)out_size; (void)ws_size;
#define IN(i) ((const float*)d_in[(i)])

    float* out = (float*)d_out;
    float* ws  = (float*)d_ws;

    // ---- f32 region ----
    float* zk   = ws;              // 4096
    float* zc   = ws + 4096;       // 512
    float* zall = ws + 4608;       // 51200  [t][b][l]

    // ---- f16 region ----
    _Float16* hb  = (_Float16*)(ws + 65536);
    _Float16* wd1 = hb;                 // 32*25*32   = 25600
    _Float16* wd2 = wd1 + 25600;        // 64*25*32   = 51200
    _Float16* wd3 = wd2 + 51200;        // 128*25*64  = 204800
    _Float16* wsx1 = wd3 + 204800;      // 25600
    _Float16* wsx2 = wsx1 + 25600;      // 51200
    _Float16* wsx3 = wsx2 + 51200;      // 204800
    _Float16* wt1 = wsx3 + 204800;      // 64*16*128  = 131072
    _Float16* wt2 = wt1 + 131072;       // 32*16*64   = 32768
    _Float16* ewT = wt2 + 32768;        // 8192*32    = 262144
    _Float16* zf  = ewT + 262144;       // 3200*32    = 102400
    _Float16* hbig = zf + 102400;

    // encoder activations (dead before emission; overlap e1p slot)
    _Float16* d1in = hbig;              // 256*36*36*32 = 10,616,832
    _Float16* d12  = d1in + 10616832;   // 256*20*20*32 =  3,276,800
    _Float16* d23  = d12 + 3276800;     // 256*12*12*64 =  2,359,296
    _Float16* d3o  = d23 + 2359296;     // 256*16*128   =    524,288
    _Float16* s1in = d3o + 524288;      // 32*36*36*32  =  1,327,104
    _Float16* s12  = s1in + 1327104;    // 32*20*20*32  =    409,600
    _Float16* s23  = s12 + 409600;      // 32*12*12*64  =    294,912
    _Float16* s3o  = s23 + 294912;      // 32*16*128    =     65,536
    // emission (stream-ordered reuse): e1p slot overlays encoder acts;
    // h0p shares the e2p slot (dead before e2p is written).
    _Float16* e1p = hbig;               // 3200*18*18*64 =  66,355,200
    _Float16* e2p = e1p + 66355200;     // 3200*34*34*32 = 118,374,400
    _Float16* h0p = e2p;                // 3200*10*10*128 = 40,960,000

    // ---- zero halos / padded buffers (whole-buffer memsets, cheap vs 23 TB/s) --
    hipMemsetAsync(d1in, 0, (size_t)10616832 * 2, stream);
    hipMemsetAsync(d12,  0, (size_t)3276800 * 2, stream);
    hipMemsetAsync(d23,  0, (size_t)2359296 * 2, stream);
    hipMemsetAsync(s1in, 0, (size_t)1327104 * 2, stream);
    hipMemsetAsync(s12,  0, (size_t)409600 * 2, stream);
    hipMemsetAsync(s23,  0, (size_t)294912 * 2, stream);
    hipMemsetAsync(h0p,  0, (size_t)40960000 * 2, stream);

    // ---- weight packing ----
    pack_w_kernel<<<100, 256, 0, stream>>>(IN(P_DW1), wd1, 32, 20, 32, 25);
    pack_w_kernel<<<200, 256, 0, stream>>>(IN(P_DW2), wd2, 64, 32, 32, 25);
    pack_w_kernel<<<800, 256, 0, stream>>>(IN(P_DW3), wd3, 128, 64, 64, 25);
    pack_w_kernel<<<100, 256, 0, stream>>>(IN(P_IW1), wsx1, 32, 5, 32, 25);
    pack_w_kernel<<<200, 256, 0, stream>>>(IN(P_IW2), wsx2, 64, 32, 32, 25);
    pack_w_kernel<<<800, 256, 0, stream>>>(IN(P_IW3), wsx3, 128, 64, 64, 25);
    pack_w_kernel<<<512, 256, 0, stream>>>(IN(P_ET1), wt1, 64, 128, 128, 16);
    pack_w_kernel<<<128, 256, 0, stream>>>(IN(P_ET2), wt2, 32, 64, 64, 16);
    pack_ewin_kernel<<<1024, 256, 0, stream>>>(IN(P_EWIN), ewT);

    // ---- activation packing ----
    pack_act_kernel<<<32768, 256, 0, stream>>>(IN(IN_D), 20 * 1024, d1in,
                                               256, 20, 32, 32, 2);
    pack_act_kernel<<<4096, 256, 0, stream>>>(IN(IN_X), 100 * 1024, s1in,
                                              32, 5, 32, 32, 2);

    // ---- domain encoder ----
    conv5s2_kernel<<<dim3(512, 1), 256, 0, stream>>>(d1in, wd1, IN(P_DB1), d12,
                                                     256, 32, 32, 32, 2, 0);
    conv5s2_kernel<<<dim3(128, 2), 256, 0, stream>>>(d12, wd2, IN(P_DB2), d23,
                                                     256, 32, 16, 64, 2, 0);
    conv5s2_kernel<<<dim3(32, 4), 256, 0, stream>>>(d23, wd3, IN(P_DB3), d3o,
                                                    256, 64, 8, 128, 0, 1);
    domain_head_kernel<<<256, 128, 0, stream>>>(d3o, IN(P_DWO), IN(P_DBO), zk);
    zc_kernel<<<1, 512, 0, stream>>>(zk, zc);

    // ---- state encoder ----
    conv5s2_kernel<<<dim3(64, 1), 256, 0, stream>>>(s1in, wsx1, IN(P_IB1), s12,
                                                    32, 32, 32, 32, 2, 0);
    conv5s2_kernel<<<dim3(16, 2), 256, 0, stream>>>(s12, wsx2, IN(P_IB2), s23,
                                                    32, 32, 16, 64, 2, 0);
    conv5s2_kernel<<<dim3(4, 4), 256, 0, stream>>>(s23, wsx3, IN(P_IB3), s3o,
                                                   32, 64, 8, 128, 0, 1);
    state_head_kernel<<<32, 128, 0, stream>>>(s3o, IN(P_IWMU), IN(P_IBMU),
                                              IN(P_IWVAR), IN(P_IBVAR),
                                              out + 3276800, out + 3277312, zall);

    // ---- sequential transition ----
    transition_kernel<<<1, 256, 0, stream>>>(
        IN(P_WG1), IN(P_BG1), IN(P_WG2), IN(P_BG2),
        IN(P_WP1), IN(P_BP1), IN(P_WP2), IN(P_BP2),
        IN(P_WZ), IN(P_BZ), zc, zall);

    // ---- emission decoder ----
    hipMemsetAsync(e1p, 0, (size_t)66355200 * 2, stream);   // encoder now dead
    pack_z_kernel<<<400, 256, 0, stream>>>(zall, zf);
    lin_kernel<<<dim3(25, 256), 256, 0, stream>>>(zf, ewT, IN(P_EBIN), h0p);
    convt_kernel<<<dim3(1600, 2, 4), 256, 0, stream>>>(h0p, wt1, IN(P_EB1), e1p,
                                                       3200, 128, 8, 64);
    hipMemsetAsync(e2p, 0, (size_t)118374400 * 2, stream);  // h0p now dead
    convt_kernel<<<dim3(6400, 1, 4), 256, 0, stream>>>(e1p, wt2, IN(P_EB2), e2p,
                                                       3200, 64, 16, 32);
    outconv_kernel<<<12800, 256, 0, stream>>>(e2p, IN(P_EO), IN(P_EBO), out);
#undef IN
}
